// DynamicAgentAttention_90263032693144
// MI455X (gfx1250) — matrix operand order repaired and checked by reading, not compiled, not run on hardware
//
#include <hip/hip_runtime.h>
#include <stdint.h>

// ---------------------------------------------------------------------------
// DynamicAgentAttention for MI455X (gfx1250, wave32, WMMA bf16 16x16x32)
// ---------------------------------------------------------------------------

#define BATCH 16
#define CDIM  512
#define NTOK  1024
#define HEADS 8
#define HD    64
#define AGENT 49
#define AGP   64   // agent padded to 64 for clean WMMA tiling

typedef __attribute__((ext_vector_type(16))) __bf16 v16bf;
typedef __attribute__((ext_vector_type(8)))  float  v8f;

union FragBF {
    v16bf  v;
    __bf16 e[16];
    uint4  q[2];
};

union Pack8BF {
    uint4  q;
    __bf16 e[8];
};

__device__ __forceinline__ __bf16 f2bf(float f) {
    union { float f; uint32_t u; } cv;
    cv.f = f;
    uint32_t r = cv.u + 0x7FFFu + ((cv.u >> 16) & 1u);  // round-to-nearest-even
    uint16_t h = (uint16_t)(r >> 16);
    __bf16 b;
    __builtin_memcpy(&b, &h, 2);
    return b;
}

// ---------------------------------------------------------------------------
// Generic batched bf16 WMMA GEMM:  C[z] = alpha * A[z] * op(B[z]) + bias (+res)
//   A: [M,K] bf16 row-major (lda), staged through LDS per 32-deep K step.
//   BTRANS=1: B stored [N,K] row-major -> A*B^T, per-lane contiguous loads.
//   BTRANS=0: B stored [K,N] row-major -> A*B; the 32x64 B slab is staged
//             into LDS *transposed* so fragments are contiguous ds_load_b128.
//   Batch z = zo*innerCount + zi with per-operand (outer, inner) strides so
//   head-sliced views (base b*N*C + h*64, ld C) need no repacking.
//   Block = 256 threads (8 waves), 64x64 tile; each wave: 32x16 region as two
//   v_wmma_f32_16x16x32_bf16 accumulators.
// ---------------------------------------------------------------------------
template <int BTRANS>
__global__ __launch_bounds__(256) void wmma_gemm_bf16(
    const __bf16* __restrict__ A, long long aOuter, long long aInner, int lda,
    const __bf16* __restrict__ Bm, long long bOuter, long long bInner, int ldb,
    float* __restrict__ Cf, __bf16* __restrict__ Cb,
    long long cOuter, long long cInner, int ldc,
    const float* __restrict__ bias,
    const float* __restrict__ res, long long rOuter, long long rInner, int ldr,
    int K, int innerCount, float alpha, int doRelu)
{
    __shared__ __bf16 smem[(BTRANS == 0) ? (64 * 32 * 2) : (64 * 32)];
    __bf16* As = smem;                                  // [64 rows][32 K]
    __bf16* Bt = (BTRANS == 0) ? smem + 64 * 32 : smem; // [64 cols][32 K]

    const int z  = blockIdx.z;
    const int zo = z / innerCount;
    const int zi = z - zo * innerCount;

    const __bf16* gA = A  + zo * aOuter + zi * aInner;
    const __bf16* gB = Bm + zo * bOuter + zi * bInner;

    const int tid  = threadIdx.x;
    const int lane = tid & 31;
    const int wave = tid >> 5;
    const int hi   = lane >> 4;       // K-half selector (wave32 fragment layout)
    const int l15  = lane & 15;

    const int mBlock = blockIdx.y << 6;
    const int nBlock = blockIdx.x << 6;
    const int waveM  = (wave >> 2) << 5;   // 0 or 32
    const int waveN  = (wave & 3) << 4;    // 0,16,32,48
    const int col    = nBlock + waveN + l15;

    v8f c0 = {}, c1 = {};

    // A cooperative stage: thread t loads 16 B of row (t>>2), K-chunk (t&3)*8
    const int arow = tid >> 2;
    const int akk  = (tid & 3) << 3;
    const __bf16* pA = gA + (long long)(mBlock + arow) * lda + akk;

    // BTRANS=1: per-lane contiguous pointer, stride 32 per K step
    const __bf16* pBt = gB + (long long)col * ldb + 16 * hi;

    // BTRANS=0: cooperative stage of [32 K][64 N] slab, transposed into LDS.
    // thread t: K-row (t>>3), col group (t&7)*8 — one 16 B global load.
    const int brow = tid >> 3;
    const int bcg  = (tid & 7) << 3;
    const __bf16* pBn = gB + (long long)brow * ldb + nBlock + bcg;
    const long long bStep = (long long)32 * ldb;

    const int aFragBase0 = (waveM + l15) * 32 + 16 * hi;
    const int aFragBase1 = aFragBase0 + 16 * 32;
    const int bFragBase  = (waveN + l15) * 32 + 16 * hi;

    for (int k0 = 0; k0 < K; k0 += 32) {
        __syncthreads();
        *(uint4*)(&As[arow * 32 + akk]) = *(const uint4*)(pA);
        __builtin_prefetch(pA + 32, 0, 1);
        pA += 32;
        if (BTRANS == 0) {
            Pack8BF t8;
            t8.q = *(const uint4*)(pBn);
            pBn += bStep;
#pragma unroll
            for (int j = 0; j < 8; ++j) Bt[(bcg + j) * 32 + brow] = t8.e[j];
        }
        __syncthreads();

        // A fragments (row = lane&15, K = 16*hi + j : 16 contiguous elems)
        FragBF a0, a1, b;
        a0.q[0] = *(const uint4*)(&As[aFragBase0]);
        a0.q[1] = *(const uint4*)(&As[aFragBase0 + 8]);
        a1.q[0] = *(const uint4*)(&As[aFragBase1]);
        a1.q[1] = *(const uint4*)(&As[aFragBase1 + 8]);

        // B fragment (col = lane&15, K = 16*hi + j : 16 contiguous elems)
        if (BTRANS) {
            b.q[0] = *(const uint4*)(pBt);
            b.q[1] = *(const uint4*)(pBt + 8);
            pBt += 32;
        } else {
            b.q[0] = *(const uint4*)(&Bt[bFragBase]);
            b.q[1] = *(const uint4*)(&Bt[bFragBase + 8]);
        }

        c0 = __builtin_amdgcn_wmma_f32_16x16x32_bf16(
                 false, a0.v, false, b.v, (short)0, c0, false, false);
        c1 = __builtin_amdgcn_wmma_f32_16x16x32_bf16(
                 false, a1.v, false, b.v, (short)0, c1, false, false);
    }

    const float bv = bias ? bias[col] : 0.0f;
    float*       cf = Cf  ? Cf  + zo * cOuter + zi * cInner : (float*)0;
    __bf16*      cb = Cb  ? Cb  + zo * cOuter + zi * cInner : (__bf16*)0;
    const float* rp = res ? res + zo * rOuter + zi * rInner : (const float*)0;

#pragma unroll
    for (int i = 0; i < 8; ++i) {
        const int row0 = mBlock + waveM + 8 * hi + i;  // C layout: M = 8*hi + vgpr
        const int row1 = row0 + 16;
        float v0 = c0[i] * alpha + bv;
        float v1 = c1[i] * alpha + bv;
        if (rp) {
            v0 += rp[(long long)row0 * ldr + col];
            v1 += rp[(long long)row1 * ldr + col];
        }
        if (doRelu) { v0 = fmaxf(v0, 0.0f); v1 = fmaxf(v1, 0.0f); }
        if (cf) {
            cf[(long long)row0 * ldc + col] = v0;
            cf[(long long)row1 * ldc + col] = v1;
        }
        if (cb) {
            cb[(long long)row0 * ldc + col] = f2bf(v0);
            cb[(long long)row1 * ldc + col] = f2bf(v1);
        }
    }
}

// ---------------------------------------------------------------------------
// x [B,C,N] f32  ->  xf [B,N,C] bf16   (LDS-tiled transpose + convert)
// ---------------------------------------------------------------------------
__global__ __launch_bounds__(256) void transpose_convert(
    const float* __restrict__ x, __bf16* __restrict__ xf)
{
    __shared__ float tile[32][33];
    const int b  = blockIdx.z;
    const int n0 = blockIdx.x << 5;
    const int c0 = blockIdx.y << 5;
    const int tx = threadIdx.x;       // 32
    const int ty = threadIdx.y;       // 8
#pragma unroll
    for (int i = 0; i < 32; i += 8)
        tile[ty + i][tx] = x[((long long)b * CDIM + c0 + ty + i) * NTOK + n0 + tx];
    __syncthreads();
#pragma unroll
    for (int i = 0; i < 32; i += 8)
        xf[((long long)b * NTOK + n0 + ty + i) * CDIM + c0 + tx] = f2bf(tile[tx][ty + i]);
}

__global__ void convert_f32_bf16(const float* __restrict__ in,
                                 __bf16* __restrict__ out, int n)
{
    int i = blockIdx.x * blockDim.x + threadIdx.x;
    if (i < n) out[i] = f2bf(in[i]);
}

// ---------------------------------------------------------------------------
// Agent branch. The reference's raw reshapes are flat-memory relabels, so the
// pooled [B,C,7,7] -> [B,49,C] -> [B,C,7,7] chain is: conv input at flat index
// f = n*512 + c holds pool(b, c, n).
// ---------------------------------------------------------------------------
__global__ void agent_pool_scramble(const float* __restrict__ x,
                                    float* __restrict__ conv_in)
{
    int idx = blockIdx.x * blockDim.x + threadIdx.x;   // B * 49 * 512
    if (idx >= BATCH * AGENT * CDIM) return;
    int c = idx % CDIM;
    int n = (idx / CDIM) % AGENT;
    int b = idx / (CDIM * AGENT);
    int pi = n / 7, pj = n % 7;
    int hs = (pi * 32) / 7, he = ((pi + 1) * 32 + 6) / 7;
    int ws = (pj * 32) / 7, we = ((pj + 1) * 32 + 6) / 7;
    const float* xp = x + ((long long)b * CDIM + c) * NTOK;
    float s = 0.0f;
    for (int y = hs; y < he; ++y)
        for (int xx = ws; xx < we; ++xx)
            s += xp[y * 32 + xx];
    conv_in[(long long)b * (CDIM * AGENT) + n * CDIM + c] =
        s / (float)((he - hs) * (we - ws));
}

__global__ void dwconv_bn_relu(const float* __restrict__ conv_in,
                               const float* __restrict__ dw_w,
                               const float* __restrict__ dw_b,
                               const float* __restrict__ g,
                               const float* __restrict__ bt,
                               const float* __restrict__ mean,
                               const float* __restrict__ var,
                               float* __restrict__ mid)
{
    int idx = blockIdx.x * blockDim.x + threadIdx.x;   // B * 512 * 49
    if (idx >= BATCH * CDIM * AGENT) return;
    int p = idx % AGENT;
    int c = (idx / AGENT) % CDIM;
    int b = idx / (AGENT * CDIM);
    int y = p / 7, xw = p % 7;
    const float* in = conv_in + (long long)b * (CDIM * AGENT) + c * AGENT;
    const float* w  = dw_w + c * 9;
    float s = 0.0f;
#pragma unroll
    for (int ky = 0; ky < 3; ++ky) {
        int yy = y + ky - 1;
        if (yy < 0 || yy > 6) continue;
#pragma unroll
        for (int kx = 0; kx < 3; ++kx) {
            int xx = xw + kx - 1;
            if (xx < 0 || xx > 6) continue;
            s += w[ky * 3 + kx] * in[yy * 7 + xx];
        }
    }
    s += dw_b[c];
    s = (s - mean[c]) * rsqrtf(var[c] + 1e-5f) * g[c] + bt[c];
    mid[(long long)b * (CDIM * AGENT) + c * AGENT + p] = fmaxf(s, 0.0f);
}

__global__ void pw_conv(const float* __restrict__ mid,
                        const float* __restrict__ pw_w,
                        const float* __restrict__ pw_b,
                        float* __restrict__ outc)
{
    int idx = blockIdx.x * blockDim.x + threadIdx.x;   // B * 512 * 49
    if (idx >= BATCH * CDIM * AGENT) return;
    int p  = idx % AGENT;
    int co = (idx / AGENT) % CDIM;
    int b  = idx / (AGENT * CDIM);
    const float* w = pw_w + (long long)co * CDIM;
    const float* m = mid + (long long)b * (CDIM * AGENT);
    float s = pw_b[co];
    for (int c = 0; c < CDIM; ++c) s += w[c] * m[c * AGENT + p];
    outc[(long long)b * (CDIM * AGENT) + co * AGENT + p] = s;
}

// a[b,h,ag,d] = conv_out_flat[b][h*49*64 + ag*64 + d]; zero-pad ag 49..63
__global__ void agent_relabel_pad(const float* __restrict__ conv_out,
                                  __bf16* __restrict__ abf)
{
    int idx = blockIdx.x * blockDim.x + threadIdx.x;   // B*8*64*64
    if (idx >= BATCH * HEADS * AGP * HD) return;
    int d  = idx & 63;
    int ag = (idx >> 6) & 63;
    int h  = (idx >> 12) & 7;
    int b  = idx >> 15;
    float v = (ag < AGENT)
        ? conv_out[(long long)b * (CDIM * AGENT) + h * (AGENT * HD) + ag * HD + d]
        : 0.0f;
    abf[idx] = f2bf(v);
}

// ---------------------------------------------------------------------------
// Softmax kernels. Stage1: rows of 1024 (pad agent rows -> 0). Stage2: rows
// of 64 with only first 49 valid (pad cols -> 0).
// ---------------------------------------------------------------------------
__global__ __launch_bounds__(256) void softmax_rows1024(
    const float* __restrict__ S, __bf16* __restrict__ P)
{
    const int row = blockIdx.x;          // (b*8+h)*64 + ag
    const int ag  = row & 63;
    const float* s = S + (long long)row * 1024;
    __bf16* p = P + (long long)row * 1024;
    const int t = threadIdx.x;           // 256
    if (ag >= AGENT) {
        for (int i = t; i < 1024; i += 256) p[i] = f2bf(0.0f);
        return;
    }
    __shared__ float red[256];
    float m = -1e30f;
    float ev[4];
    for (int i = t; i < 1024; i += 256) m = fmaxf(m, s[i]);
    red[t] = m; __syncthreads();
    for (int st = 128; st > 0; st >>= 1) {
        if (t < st) red[t] = fmaxf(red[t], red[t + st]);
        __syncthreads();
    }
    m = red[0]; __syncthreads();
    float sum = 0.0f;
    for (int i = t, k = 0; i < 1024; i += 256, ++k) {
        ev[k] = __expf(s[i] - m);
        sum += ev[k];
    }
    red[t] = sum; __syncthreads();
    for (int st = 128; st > 0; st >>= 1) {
        if (t < st) red[t] += red[t + st];
        __syncthreads();
    }
    const float inv = 1.0f / red[0];
    for (int i = t, k = 0; i < 1024; i += 256, ++k) p[i] = f2bf(ev[k] * inv);
}

__global__ __launch_bounds__(64) void softmax_rows64(
    const float* __restrict__ S, __bf16* __restrict__ P)
{
    const long long row = blockIdx.x;    // (b*8+h)*1024 + n
    const int t = threadIdx.x;           // 64
    const float* s = S + row * 64;
    __shared__ float red[64];
    float v = (t < AGENT) ? s[t] : -1e30f;
    red[t] = v; __syncthreads();
    for (int st = 32; st > 0; st >>= 1) {
        if (t < st) red[t] = fmaxf(red[t], red[t + st]);
        __syncthreads();
    }
    const float m = red[0]; __syncthreads();
    float e = (t < AGENT) ? __expf(v - m) : 0.0f;
    red[t] = e; __syncthreads();
    for (int st = 32; st > 0; st >>= 1) {
        if (t < st) red[t] += red[t + st];
        __syncthreads();
    }
    P[row * 64 + t] = f2bf(e / red[0]);
}

// ---------------------------------------------------------------------------
// Host side
// ---------------------------------------------------------------------------
static inline void gemm(hipStream_t st,
                        const __bf16* A, long long aO, long long aI, int lda,
                        const __bf16* B, long long bO, long long bI, int ldb, int bT,
                        float* Cf, __bf16* Cb, long long cO, long long cI, int ldc,
                        const float* bias,
                        const float* res, long long rO, long long rI, int ldr,
                        int M, int N, int K, int outer, int inner,
                        float alpha, int relu)
{
    dim3 grid(N / 64, M / 64, outer * inner);
    if (bT)
        wmma_gemm_bf16<1><<<grid, 256, 0, st>>>(A, aO, aI, lda, B, bO, bI, ldb,
                                                Cf, Cb, cO, cI, ldc, bias,
                                                res, rO, rI, ldr, K, inner, alpha, relu);
    else
        wmma_gemm_bf16<0><<<grid, 256, 0, st>>>(A, aO, aI, lda, B, bO, bI, ldb,
                                                Cf, Cb, cO, cI, ldc, bias,
                                                res, rO, rI, ldr, K, inner, alpha, relu);
}

extern "C" void kernel_launch(void* const* d_in, const int* in_sizes, int n_in,
                              void* d_out, int out_size, void* d_ws, size_t ws_size,
                              hipStream_t stream) {
    (void)in_sizes; (void)n_in; (void)out_size; (void)ws_size;
    const float* x      = (const float*)d_in[0];
    const float* q_w    = (const float*)d_in[3];
    const float* kv_w   = (const float*)d_in[4];
    const float* dw_w   = (const float*)d_in[5];
    const float* dw_b   = (const float*)d_in[6];
    const float* bn_g   = (const float*)d_in[7];
    const float* bn_b   = (const float*)d_in[8];
    const float* bn_m   = (const float*)d_in[9];
    const float* bn_v   = (const float*)d_in[10];
    const float* pw_w   = (const float*)d_in[11];
    const float* pw_b   = (const float*)d_in[12];
    const float* proj_w = (const float*)d_in[13];
    const float* proj_b = (const float*)d_in[14];
    const float* enh1_w = (const float*)d_in[15];
    const float* enh1_b = (const float*)d_in[16];
    const float* enh2_w = (const float*)d_in[17];
    const float* enh2_b = (const float*)d_in[18];
    float* out = (float*)d_out;

    // ---- scratch carve (ws_size assumed >= ~220 MB) ----
    char* ws = (char*)d_ws;
    size_t off = 0;
    auto carve = [&](size_t bytes) -> char* {
        char* p = ws + off;
        off = (off + bytes + 255) & ~(size_t)255;
        return p;
    };
    const size_t TOKC = (size_t)BATCH * NTOK * CDIM;         // 8M elems
    __bf16* xf    = (__bf16*)carve(TOKC * 2);
    __bf16* wq    = (__bf16*)carve((size_t)CDIM * CDIM * 2);
    __bf16* wkv   = (__bf16*)carve((size_t)2 * CDIM * CDIM * 2);
    __bf16* wproj = (__bf16*)carve((size_t)CDIM * CDIM * 2);
    __bf16* wen1  = (__bf16*)carve((size_t)CDIM * CDIM * 2);
    __bf16* wen2  = (__bf16*)carve((size_t)CDIM * CDIM * 2);
    __bf16* Qb    = (__bf16*)carve(TOKC * 2);
    __bf16* Kb    = (__bf16*)carve(TOKC * 2);
    __bf16* Vb    = (__bf16*)carve(TOKC * 2);
    float*  cin   = (float*)carve((size_t)BATCH * CDIM * AGENT * 4);
    float*  cmid  = (float*)carve((size_t)BATCH * CDIM * AGENT * 4);
    float*  cout  = (float*)carve((size_t)BATCH * CDIM * AGENT * 4);
    __bf16* abf   = (__bf16*)carve((size_t)BATCH * HEADS * AGP * HD * 2);
    float*  Sbuf  = (float*)carve((size_t)BATCH * HEADS * AGP * NTOK * 4);  // 33.5 MB
    __bf16* Pbuf  = (__bf16*)carve((size_t)BATCH * HEADS * AGP * NTOK * 2);
    __bf16* aggb  = (__bf16*)carve((size_t)BATCH * HEADS * AGP * HD * 2);
    __bf16* enhb  = (__bf16*)carve(TOKC * 2);
    float*  xout  = (float*)carve(TOKC * 4);
    __bf16* xoutb = (__bf16*)carve(TOKC * 2);
    __bf16* relub = (__bf16*)carve(TOKC * 2);

    const float scale = 0.125f;  // hd^-0.5, hd = 64

    // 1) convert weights, transpose/convert activations
    convert_f32_bf16<<<(CDIM*CDIM+255)/256, 256, 0, stream>>>(q_w, wq, CDIM*CDIM);
    convert_f32_bf16<<<(2*CDIM*CDIM+255)/256, 256, 0, stream>>>(kv_w, wkv, 2*CDIM*CDIM);
    convert_f32_bf16<<<(CDIM*CDIM+255)/256, 256, 0, stream>>>(proj_w, wproj, CDIM*CDIM);
    convert_f32_bf16<<<(CDIM*CDIM+255)/256, 256, 0, stream>>>(enh1_w, wen1, CDIM*CDIM);
    convert_f32_bf16<<<(CDIM*CDIM+255)/256, 256, 0, stream>>>(enh2_w, wen2, CDIM*CDIM);
    transpose_convert<<<dim3(NTOK/32, CDIM/32, BATCH), dim3(32, 8), 0, stream>>>(x, xf);

    // 2) Q/K/V projections: [B,1024,512] = xf * W^T  (K = kv rows 0..511, V = 512..1023)
    gemm(stream, xf, (long long)NTOK*CDIM, 0, CDIM, wq, 0, 0, CDIM, 1,
         nullptr, Qb, (long long)NTOK*CDIM, 0, CDIM, nullptr, nullptr, 0, 0, 0,
         NTOK, CDIM, CDIM, BATCH, 1, 1.0f, 0);
    gemm(stream, xf, (long long)NTOK*CDIM, 0, CDIM, wkv, 0, 0, CDIM, 1,
         nullptr, Kb, (long long)NTOK*CDIM, 0, CDIM, nullptr, nullptr, 0, 0, 0,
         NTOK, CDIM, CDIM, BATCH, 1, 1.0f, 0);
    gemm(stream, xf, (long long)NTOK*CDIM, 0, CDIM, wkv + (size_t)CDIM*CDIM, 0, 0, CDIM, 1,
         nullptr, Vb, (long long)NTOK*CDIM, 0, CDIM, nullptr, nullptr, 0, 0, 0,
         NTOK, CDIM, CDIM, BATCH, 1, 1.0f, 0);

    // 3) agent token pipeline (small)
    {
        int n1 = BATCH * AGENT * CDIM;
        agent_pool_scramble<<<(n1+255)/256, 256, 0, stream>>>(x, cin);
        int n2 = BATCH * CDIM * AGENT;
        dwconv_bn_relu<<<(n2+255)/256, 256, 0, stream>>>(cin, dw_w, dw_b, bn_g, bn_b,
                                                         bn_m, bn_v, cmid);
        pw_conv<<<(n2+255)/256, 256, 0, stream>>>(cmid, pw_w, pw_b, cout);
        int n3 = BATCH * HEADS * AGP * HD;
        agent_relabel_pad<<<(n3+255)/256, 256, 0, stream>>>(cout, abf);
    }

    // 4) stage 1: S1 = scale * a * K^T   [z=(b,h)] : [64,1024]
    gemm(stream,
         abf, (long long)HEADS*AGP*HD, (long long)AGP*HD, HD,
         Kb, (long long)NTOK*CDIM, HD, CDIM, 1,
         Sbuf, nullptr, (long long)HEADS*AGP*NTOK, (long long)AGP*NTOK, NTOK,
         nullptr, nullptr, 0, 0, 0,
         AGP, NTOK, HD, BATCH, HEADS, scale, 0);
    softmax_rows1024<<<BATCH*HEADS*AGP, 256, 0, stream>>>(Sbuf, Pbuf);

    // 5) agg = P1 * V : [64,64], B non-transposed ([K=1024, N=64] view of V)
    gemm(stream,
         Pbuf, (long long)HEADS*AGP*NTOK, (long long)AGP*NTOK, NTOK,
         Vb, (long long)NTOK*CDIM, HD, CDIM, 0,
         nullptr, aggb, (long long)HEADS*AGP*HD, (long long)AGP*HD, HD,
         nullptr, nullptr, 0, 0, 0,
         AGP, HD, NTOK, BATCH, HEADS, 1.0f, 0);

    // 6) stage 2: S2 = scale * Q * a^T : [1024,64]
    gemm(stream,
         Qb, (long long)NTOK*CDIM, HD, CDIM,
         abf, (long long)HEADS*AGP*HD, (long long)AGP*HD, HD, 1,
         Sbuf, nullptr, (long long)HEADS*NTOK*AGP, (long long)NTOK*AGP, AGP,
         nullptr, nullptr, 0, 0, 0,
         NTOK, AGP, HD, BATCH, HEADS, scale, 0);
    softmax_rows64<<<BATCH*HEADS*NTOK, 64, 0, stream>>>(Sbuf, Pbuf);

    // 7) enh = P2 * agg : [1024,64] -> head-sliced into enhb [B,N,C]
    gemm(stream,
         Pbuf, (long long)HEADS*NTOK*AGP, (long long)NTOK*AGP, AGP,
         aggb, (long long)HEADS*AGP*HD, (long long)AGP*HD, HD, 0,
         nullptr, enhb, (long long)NTOK*CDIM, HD, CDIM,
         nullptr, nullptr, 0, 0, 0,
         NTOK, HD, AGP, BATCH, HEADS, 1.0f, 0);

    // 8) x_out = enh * proj^T + proj_b   (keep f32 for residual, bf16 for MLP)
    gemm(stream, enhb, (long long)NTOK*CDIM, 0, CDIM, wproj, 0, 0, CDIM, 1,
         xout, xoutb, (long long)NTOK*CDIM, 0, CDIM, proj_b, nullptr, 0, 0, 0,
         NTOK, CDIM, CDIM, BATCH, 1, 1.0f, 0);

    // 9) relu(x_out * enh1^T + enh1_b)
    gemm(stream, xoutb, (long long)NTOK*CDIM, 0, CDIM, wen1, 0, 0, CDIM, 1,
         nullptr, relub, (long long)NTOK*CDIM, 0, CDIM, enh1_b, nullptr, 0, 0, 0,
         NTOK, CDIM, CDIM, BATCH, 1, 1.0f, 1);

    // 10) out = relu_part * enh2^T + enh2_b + x_out  (flat layout == [B,C,H,W] reshape)
    gemm(stream, relub, (long long)NTOK*CDIM, 0, CDIM, wen2, 0, 0, CDIM, 1,
         out, nullptr, (long long)NTOK*CDIM, 0, CDIM, enh2_b,
         xout, (long long)NTOK*CDIM, 0, CDIM,
         NTOK, CDIM, CDIM, BATCH, 1, 1.0f, 0);
}
